// VarLenSelfAttention_59459527246255
// MI455X (gfx1250) — matrix operand order repaired and checked by reading, not compiled, not run on hardware
//
#include <hip/hip_runtime.h>
#include <hip/hip_bf16.h>
#include <stdint.h>

// Problem constants (match reference setup_inputs()).
#define TT     4096   // total tokens (B*S)
#define HID    2048
#define NH     16
#define NKV    4
#define HD     128
#define SLEN   1024
#define BATCH  4
#define KVDIM  512    // NKV*HD
#define WINDOW 256

typedef __attribute__((ext_vector_type(16))) __bf16 bf16x16;
typedef __attribute__((ext_vector_type(8)))  float  f32x8;
typedef __attribute__((address_space(3))) uint16_t lds16_t;

union Frag {
  bf16x16 v;
  uint4    q[2];
  uint32_t u[8];
};

__device__ __forceinline__ uint32_t f2bf(float f) {
  uint32_t u = __builtin_bit_cast(uint32_t, f);
  return (u + 0x7fffu + ((u >> 16) & 1u)) >> 16;   // round-to-nearest-even
}
__device__ __forceinline__ uint32_t pack2(float lo, float hi) {
  return f2bf(lo) | (f2bf(hi) << 16);
}

__device__ __forceinline__ f32x8 wmma_bf16(const Frag& a, const Frag& b, f32x8 c) {
  return __builtin_amdgcn_wmma_f32_16x16x32_bf16(
      /*neg_a=*/false, a.v, /*neg_b=*/false, b.v,
      /*c_mod=*/(short)0, c, /*reuse_a=*/false, /*reuse_b=*/false);
}

// A fragment, 16x32 bf16, element (m,k) at base + m*stride + k.
// Lane L: m=L&15; L<16 holds K {0..7,16..23}, L>=16 holds K {8..15,24..31}.
__device__ __forceinline__ Frag load_a32(const uint16_t* base, int stride) {
  int l = threadIdx.x & 31, m = l & 15, h = l >> 4;
  const uint16_t* p = base + (size_t)m * stride + h * 8;
  Frag f;
  f.q[0] = *(const uint4*)p;
  f.q[1] = *(const uint4*)(p + 16);
  return f;
}
// B fragment, 32x16 bf16, element (k,n) at base + n*stride + k (B source pre-transposed).
// Lanes 0-15 hold K=0..15 of column n=L&15; lanes 16-31 hold K=16..31.
__device__ __forceinline__ Frag load_b32(const uint16_t* base, int stride) {
  int l = threadIdx.x & 31, n = l & 15, h = l >> 4;
  const uint16_t* p = base + (size_t)n * stride + h * 16;
  Frag f;
  f.q[0] = *(const uint4*)p;
  f.q[1] = *(const uint4*)(p + 8);
  return f;
}
// B fragment from LDS slab (16 rows, padded stride of 40 uint16 = 80B per row).
__device__ __forceinline__ Frag load_b_lds(const uint16_t* base) {
  int l = threadIdx.x & 31, n = l & 15, h = l >> 4;
  const uint16_t* p = base + n * 40 + h * 8;
  Frag f;
  f.q[0] = *(const uint4*)p;
  f.q[1] = *(const uint4*)(p + 16);
  return f;
}

// ---------------------------------------------------------------- converters
__global__ void cvt_f32_bf16(const float* __restrict__ in, uint16_t* __restrict__ out, int n) {
  int i = blockIdx.x * blockDim.x + threadIdx.x;
  if (i < n) out[i] = (uint16_t)f2bf(in[i]);
}
// in: [K][N] row-major f32  ->  out: [N][K] bf16
__global__ void trans_f32_bf16(const float* __restrict__ in, uint16_t* __restrict__ out,
                               int K, int N) {
  int i = blockIdx.x * blockDim.x + threadIdx.x;
  if (i >= K * N) return;
  int k = i / N, n = i - k * N;
  out[(size_t)n * K + k] = (uint16_t)f2bf(in[i]);
}

// ---------------------------------------------------------------- GEMM
// C[M,N] = A_bf16[M,K] * Bt_bf16[N,K]^T.
// Workgroup = 8 waves -> 128M x 64N macro-tile. The 64x32 B slab per K-step is
// staged once per workgroup into LDS with global_load_async_to_lds_b128
// (ASYNCcnt), double-buffered so the next slab's DMA overlaps current WMMAs.
// Each wave computes a 16M x 64N strip: per K-step 1 A fragment (2x b128) and
// 4 B fragments read from LDS (ds_load_b128), 4 WMMAs.
// MODE 0: f32 row-major store. MODE 1: bf16 store transposed as [b][n][s].
template <int MODE>
__global__ __launch_bounds__(256) void gemm_bf16_kernel(
    const uint16_t* __restrict__ A, const uint16_t* __restrict__ Bt,
    void* __restrict__ Cout, int M, int N, int K) {
  __shared__ uint16_t ldsB[2][64][40];   // 2 x 64 rows x (32 data + 8 pad) bf16

  int nbN = N >> 6;
  int mb  = blockIdx.x / nbN;
  int ng  = blockIdx.x - mb * nbN;
  int w   = threadIdx.x >> 5;
  int mt  = mb * 8 + w;                  // this wave's 16-row M tile

  // async B staging: thread t copies 16B: row = t/4 (0..63), segment = t%4
  int tid = threadIdx.x;
  int row = tid >> 2, seg = tid & 3;
  const uint16_t* bsrc = Bt + (size_t)(ng * 64 + row) * K + seg * 8;

  auto issue_copy = [&](int kc, int buf) {
    uint64_t ga = (uint64_t)(uintptr_t)(bsrc + kc);
    uint32_t la = (uint32_t)(uintptr_t)(lds16_t*)&ldsB[buf][row][seg * 8];
    asm volatile("global_load_async_to_lds_b128 %0, %1, off"
                 :: "v"(la), "v"(ga) : "memory");
  };

  f32x8 acc[4] = {};
  const uint16_t* arow = A + (size_t)mt * 16 * K;

  issue_copy(0, 0);
  int nk = K >> 5;
  for (int i = 0; i < nk; ++i) {
    asm volatile("s_wait_asynccnt 0x0" ::: "memory");
    __syncthreads();                      // slab i visible; buf (i+1)&1 free
    if (i + 1 < nk) issue_copy((i + 1) << 5, (i + 1) & 1);
    int buf = i & 1;
    Frag a = load_a32(arow + (i << 5), K);
#pragma unroll
    for (int j = 0; j < 4; ++j) {
      Frag b = load_b_lds(&ldsB[buf][j * 16][0]);
      acc[j] = wmma_bf16(a, b, acc[j]);
    }
  }

  int l = threadIdx.x & 31, n0 = l & 15, h = l >> 4;
  if (MODE == 0) {
    float* C = (float*)Cout;
#pragma unroll
    for (int j = 0; j < 4; ++j) {
      int n = ng * 64 + j * 16 + n0;
#pragma unroll
      for (int r = 0; r < 8; ++r)
        C[(size_t)(mt * 16 + r + 8 * h) * N + n] = acc[j][r];
    }
  } else {
    // tokens of this tile are contiguous along s; 8 rows -> one 16B bf16 store
    uint16_t* C = (uint16_t*)Cout;
    int t0 = mt * 16, bb = t0 >> 10, s0 = t0 & (SLEN - 1);
#pragma unroll
    for (int j = 0; j < 4; ++j) {
      int n = ng * 64 + j * 16 + n0;
      uint4 wv;
      uint32_t* wp = (uint32_t*)&wv;
#pragma unroll
      for (int r = 0; r < 4; ++r) wp[r] = pack2(acc[j][2 * r], acc[j][2 * r + 1]);
      *(uint4*)(C + ((size_t)bb * N + n) * SLEN + s0 + 8 * h) = wv;
    }
  }
}

// ---------------------------------------------------------------- RoPE (interleaved)
__global__ void rope_q_kernel(const float* __restrict__ qraw, const int* __restrict__ pos,
                              uint16_t* __restrict__ qb) {
  int idx = blockIdx.x * blockDim.x + threadIdx.x;       // t*1024 + h*64 + j
  if (idx >= TT * (NH * HD / 2)) return;
  int t = idx >> 10, rem = idx & 1023, hh = rem >> 6, j = rem & 63;
  float p = (float)pos[t];
  float freq = __powf(10000.0f, -(float)j * (1.0f / 64.0f));
  float c, s;
  __sincosf(p * freq, &c, &s);
  size_t base = (size_t)t * HID + hh * HD + 2 * j;
  float xe = qraw[base], xo = qraw[base + 1];
  *(uint32_t*)(qb + base) = pack2(xe * c - xo * s, xe * s + xo * c);
}
__global__ void rope_k_kernel(const float* __restrict__ kraw, const int* __restrict__ pos,
                              uint16_t* __restrict__ kb) {
  int idx = blockIdx.x * blockDim.x + threadIdx.x;       // t*256 + kv*64 + j
  if (idx >= TT * (KVDIM / 2)) return;
  int t = idx >> 8, rem = idx & 255, hh = rem >> 6, j = rem & 63;
  float p = (float)pos[t];
  float freq = __powf(10000.0f, -(float)j * (1.0f / 64.0f));
  float c, s;
  __sincosf(p * freq, &c, &s);
  size_t base = (size_t)t * KVDIM + hh * HD + 2 * j;
  float xe = kraw[base], xo = kraw[base + 1];
  *(uint32_t*)(kb + base) = pack2(xe * c - xo * s, xe * s + xo * c);
}

// ---------------------------------------------------------------- attention
// One wave per (batch, head, 16-query block). Computes scores^T = K_tile x Q^T
// so softmax stats are lane-local (column = query), online softmax over 32-key
// chunks within the causal sliding window, probs re-packed lane-locally into an
// A fragment, then P x V via WMMA (V stored transposed bf16 [b][kv*128+d][s]).
__global__ __launch_bounds__(32) void attn_kernel(
    const uint16_t* __restrict__ qb, const uint16_t* __restrict__ kb,
    const uint16_t* __restrict__ vt, uint16_t* __restrict__ attnb) {
  int unit = blockIdx.x;                 // b*NH*64 + h*64 + qblk
  int qblk = unit & 63;
  int h    = (unit >> 6) & 15;
  int b    = unit >> 10;
  int kv   = h >> 2;

  int lane = threadIdx.x & 31;
  int qcol = lane & 15;                  // query column (score tiles) / dim column (acc tiles)
  int half = lane >> 4;
  int tb   = b * SLEN + qblk * 16;       // first token of query tile

  // Q as B-operand: 4 fragments covering head_dim=128 (K=32 each)
  Frag qf[4];
  {
    const uint16_t* qp = qb + (size_t)(tb + qcol) * HID + h * HD;
#pragma unroll
    for (int c = 0; c < 4; ++c) {
      const uint16_t* p = qp + c * 32 + half * 8;
      qf[c].q[0] = *(const uint4*)p;
      qf[c].q[1] = *(const uint4*)(p + 16);
    }
  }

  f32x8 acc[8] = {};                     // 8 N-tiles of 16 dims = 128
  float m_run = -1e30f, s_run = 0.0f;
  int   i_q = qblk * 16 + qcol;          // query position within sequence

  const uint16_t* kbp = kb + (size_t)b * SLEN * KVDIM + kv * HD;
  const uint16_t* vbp = vt + ((size_t)b * KVDIM + kv * HD) * SLEN;

  int kstart = max(0, qblk * 16 - WINDOW) & ~31;
  int kend   = qblk * 16 + 16;
  const float scale = 0.08838834764831845f;   // 1/sqrt(128)

  for (int k0 = kstart; k0 < kend; k0 += 32) {
    // scores^T: two 16-key tiles, A = K rows, B = Q
    f32x8 sc[2] = {};
#pragma unroll
    for (int tI = 0; tI < 2; ++tI) {
      const uint16_t* ap = kbp + (size_t)(k0 + tI * 16 + qcol) * KVDIM;
#pragma unroll
      for (int c = 0; c < 4; ++c) {
        Frag a;
        const uint16_t* p = ap + c * 32 + half * 8;
        a.q[0] = *(const uint4*)p;
        a.q[1] = *(const uint4*)(p + 16);
        sc[tI] = wmma_bf16(a, qf[c], sc[tI]);
      }
    }
    // mask (causal + left window) + scale; lane holds col q=qcol, rows key
    float lmax = -1e30f;
#pragma unroll
    for (int tI = 0; tI < 2; ++tI)
#pragma unroll
      for (int r = 0; r < 8; ++r) {
        int key = k0 + tI * 16 + r + 8 * half;
        float v = sc[tI][r] * scale;
        bool ok = (key <= i_q) && (i_q - key <= WINDOW);
        v = ok ? v : -1e30f;
        sc[tI][r] = v;
        lmax = fmaxf(lmax, v);
      }
    lmax = fmaxf(lmax, __shfl_xor(lmax, 16, 32));
    float m_new = fmaxf(m_run, lmax);
    float corr  = __expf(m_run - m_new);

    // rescale accumulators; acc row = r + 8*half, its corr lives on lane (r+8*half)
    float cr[8];
#pragma unroll
    for (int r = 0; r < 8; ++r) cr[r] = __shfl(corr, r + 8 * half, 32);
#pragma unroll
    for (int j = 0; j < 8; ++j)
#pragma unroll
      for (int r = 0; r < 8; ++r) acc[j][r] *= cr[r];

    // probs + lane-local repack into A fragment (D-layout -> A-layout, no shuffles)
    float lsum = 0.0f;
    Frag pa;
#pragma unroll
    for (int tI = 0; tI < 2; ++tI) {
      float p0[8];
#pragma unroll
      for (int r = 0; r < 8; ++r) {
        float e = __expf(sc[tI][r] - m_new);
        p0[r] = e;
        lsum += e;
      }
#pragma unroll
      for (int jj = 0; jj < 4; ++jj) pa.u[tI * 4 + jj] = pack2(p0[2 * jj], p0[2 * jj + 1]);
    }
    lsum += __shfl_xor(lsum, 16, 32);
    s_run = s_run * corr + lsum;
    m_run = m_new;

    // P x V: B from transposed V (contiguous in keys)
#pragma unroll
    for (int j = 0; j < 8; ++j) {
      const uint16_t* p = vbp + (size_t)(j * 16 + qcol) * SLEN + k0 + half * 8;
      Frag bv;
      bv.q[0] = *(const uint4*)p;
      bv.q[1] = *(const uint4*)(p + 16);
      acc[j] = wmma_bf16(pa, bv, acc[j]);
    }
  }

  // normalize rows by softmax sum and store bf16 [token][h*128+d]
  float inv = 1.0f / s_run;
  float rs[8];
#pragma unroll
  for (int r = 0; r < 8; ++r) rs[r] = __shfl(inv, r + 8 * half, 32);
#pragma unroll
  for (int j = 0; j < 8; ++j)
#pragma unroll
    for (int r = 0; r < 8; ++r) {
      int tok = tb + r + 8 * half;
      attnb[(size_t)tok * HID + h * HD + j * 16 + qcol] = (uint16_t)f2bf(acc[j][r] * rs[r]);
    }
}

// ---------------------------------------------------------------- launch
extern "C" void kernel_launch(void* const* d_in, const int* in_sizes, int n_in,
                              void* d_out, int out_size, void* d_ws, size_t ws_size,
                              hipStream_t stream) {
  (void)in_sizes; (void)n_in; (void)out_size; (void)ws_size;

  const float* hid = (const float*)d_in[0];
  const float* wq  = (const float*)d_in[1];
  const float* wk  = (const float*)d_in[2];
  const float* wv  = (const float*)d_in[3];
  const float* wo  = (const float*)d_in[4];
  const int*   pos = (const int*)d_in[6];

  // workspace layout (bytes)
  char* ws = (char*)d_ws;
  size_t off = 0;
  uint16_t* xb   = (uint16_t*)(ws + off); off += (size_t)TT * HID * 2;       // hidden bf16
  uint16_t* wqt  = (uint16_t*)(ws + off); off += (size_t)HID * HID * 2;      // wq^T bf16
  uint16_t* wkt  = (uint16_t*)(ws + off); off += (size_t)KVDIM * HID * 2;    // wk^T bf16
  uint16_t* wvt  = (uint16_t*)(ws + off); off += (size_t)KVDIM * HID * 2;    // wv^T bf16
  uint16_t* wot  = (uint16_t*)(ws + off); off += (size_t)HID * HID * 2;      // wo^T bf16
  float*    qraw = (float*)(ws + off);    off += (size_t)TT * HID * 4;       // Q pre-RoPE
  float*    kraw = (float*)(ws + off);    off += (size_t)TT * KVDIM * 4;     // K pre-RoPE
  uint16_t* qbuf = (uint16_t*)(ws + off); off += (size_t)TT * HID * 2;       // Q bf16 (RoPE'd)
  uint16_t* kbuf = (uint16_t*)(ws + off); off += (size_t)TT * KVDIM * 2;     // K bf16 (RoPE'd)
  uint16_t* vtb  = (uint16_t*)(ws + off); off += (size_t)BATCH * KVDIM * SLEN * 2; // V^T bf16
  uint16_t* attb = (uint16_t*)(ws + off); off += (size_t)TT * HID * 2;       // attn bf16

  const int TPB = 256;
  // 1) precision conversion / weight transposition
  cvt_f32_bf16<<<(TT * HID + TPB - 1) / TPB, TPB, 0, stream>>>(hid, xb, TT * HID);
  trans_f32_bf16<<<(HID * HID + TPB - 1) / TPB, TPB, 0, stream>>>(wq, wqt, HID, HID);
  trans_f32_bf16<<<(HID * KVDIM + TPB - 1) / TPB, TPB, 0, stream>>>(wk, wkt, HID, KVDIM);
  trans_f32_bf16<<<(HID * KVDIM + TPB - 1) / TPB, TPB, 0, stream>>>(wv, wvt, HID, KVDIM);
  trans_f32_bf16<<<(HID * HID + TPB - 1) / TPB, TPB, 0, stream>>>(wo, wot, HID, HID);

  // 2) QKV projections (WMMA bf16, f32 accumulate; async-LDS staged B)
  gemm_bf16_kernel<0><<<(TT / 128) * (HID / 64), TPB, 0, stream>>>(xb, wqt, qraw, TT, HID, HID);
  gemm_bf16_kernel<0><<<(TT / 128) * (KVDIM / 64), TPB, 0, stream>>>(xb, wkt, kraw, TT, KVDIM, HID);
  gemm_bf16_kernel<1><<<(TT / 128) * (KVDIM / 64), TPB, 0, stream>>>(xb, wvt, vtb, TT, KVDIM, HID);

  // 3) RoPE -> bf16 Q/K
  rope_q_kernel<<<(TT * (NH * HD / 2) + TPB - 1) / TPB, TPB, 0, stream>>>(qraw, pos, qbuf);
  rope_k_kernel<<<(TT * (KVDIM / 2) + TPB - 1) / TPB, TPB, 0, stream>>>(kraw, pos, kbuf);

  // 4) sliding-window GQA attention, one wave per (b, h, 16-query block)
  attn_kernel<<<BATCH * NH * (SLEN / 16), 32, 0, stream>>>(qbuf, kbuf, vtb, attb);

  // 5) output projection -> f32 d_out
  gemm_bf16_kernel<0><<<(TT / 128) * (HID / 64), TPB, 0, stream>>>(attb, wot, (float*)d_out, TT, HID, HID);
}
